// Qwen2MoeSparseMoeBlock_75960791597568
// MI455X (gfx1250) — matrix operand order repaired and checked by reading, not compiled, not run on hardware
//
#include <hip/hip_runtime.h>
#include <hip/hip_bf16.h>

// ---------------------------------------------------------------------------
// Qwen2 sparse-MoE block for MI455X (gfx1250, wave32, WMMA).
//
// Roofline: 71 GFLOP, 2.2 GB fp32 weights -> 94us HBM floor (23.3 TB/s).
// fp32 WMMA (16x16x4) machine balance (~13-26 FLOP/B) is borderline vs the
// experts' 17 FLOP/B, so GEMMs run as split-bf16 (x = hi + lo, 3-term
// product) on V_WMMA_F32_16X16X32_BF16 with f32 accumulate: ~5e-5 relative
// error (fp32-grade) and solidly memory-bound. hi/lo packing lowers to
// v_fma_mix_f32_bf16 + v_perm_b32 (~1.5 VALU/element).
// ---------------------------------------------------------------------------

typedef float    v4f   __attribute__((ext_vector_type(4)));
typedef float    v8f   __attribute__((ext_vector_type(8)));
typedef unsigned v2u   __attribute__((ext_vector_type(2)));
typedef unsigned v4u   __attribute__((ext_vector_type(4)));
typedef __bf16   v16bf __attribute__((ext_vector_type(16)));

constexpr int kH     = 2048;
constexpr int kI     = 1408;
constexpr int kSI    = 5632;
constexpr int kE     = 60;
constexpr int kT     = 512;           // B*S tokens
constexpr int kTopK  = 4;
constexpr int kChunk = 1408;          // I-chunk held in LDS act buffer
constexpr int kThreads = 512;         // 16 waves
constexpr int kWaves   = 16;
constexpr int kNT      = 8;           // 16-col output tiles per wave (8*16*16=2048)
constexpr int kXStrH = kH + 8;        // ushort row stride for X hi/lo (bank skew)
constexpr int kAStrH = kChunk + 8;    // ushort row stride for act hi/lo
constexpr int kSmemBytes = (2 * 16 * kXStrH + 2 * 16 * kAStrH) * 2;  // ~217 KB

// ---------------------------------------------------------------------------
// bf16x2 split: hi = truncate(x), lo = truncate(x - hi); x - hi is exact in
// fp32, total product error ~3*2^-16. V_PERM_B32 packs two hi16s per instr.
__device__ __forceinline__ void split2(float x0, float x1,
                                       unsigned& h, unsigned& l) {
  unsigned u0 = __float_as_uint(x0), u1 = __float_as_uint(x1);
  h = __builtin_amdgcn_perm(u1, u0, 0x07060302u);   // {hi16(u1), hi16(u0)}
  float r0 = x0 - __uint_as_float(u0 & 0xffff0000u);
  float r1 = x1 - __uint_as_float(u1 & 0xffff0000u);
  l = __builtin_amdgcn_perm(__float_as_uint(r1), __float_as_uint(r0),
                            0x07060302u);
}

union BV {
  v16bf v;
  unsigned u[8];
};

// Split 16 contiguous fp32 (16B-aligned) into hi/lo v16bf (B-operand layout:
// lane covers 16 consecutive K, element order linear).
__device__ __forceinline__ void split16(const float* __restrict__ p,
                                        BV& hi, BV& lo) {
  v4f w0 = *(const v4f*)(p);
  v4f w1 = *(const v4f*)(p + 4);
  v4f w2 = *(const v4f*)(p + 8);
  v4f w3 = *(const v4f*)(p + 12);
  split2(w0[0], w0[1], hi.u[0], lo.u[0]);
  split2(w0[2], w0[3], hi.u[1], lo.u[1]);
  split2(w1[0], w1[1], hi.u[2], lo.u[2]);
  split2(w1[2], w1[3], hi.u[3], lo.u[3]);
  split2(w2[0], w2[1], hi.u[4], lo.u[4]);
  split2(w2[2], w2[3], hi.u[5], lo.u[5]);
  split2(w3[0], w3[1], hi.u[6], lo.u[6]);
  split2(w3[2], w3[3], hi.u[7], lo.u[7]);
}

__device__ __forceinline__ v8f wmma_bf(v16bf a, v16bf b, v8f c) {
  return __builtin_amdgcn_wmma_f32_16x16x32_bf16(false, a, false, b,
                                                 (short)0, c, false, false);
}

// Fast sigmoid: v_rcp_f32 (~1 ulp) is far below the bf16x2 noise floor.
__device__ __forceinline__ float fast_sigmoid(float z) {
  return __builtin_amdgcn_rcpf(1.0f + __expf(-z));
}

// ---------------------------------------------------------------------------
__global__ void zero_kernel(float* __restrict__ out, int* __restrict__ counts) {
  int i = blockIdx.x * blockDim.x + threadIdx.x;
  if (i < kT * kH) out[i] = 0.0f;
  if (i < kE) counts[i] = 0;
}

// ---------------------------------------------------------------------------
// Router: logits -> softmax -> top-4 -> per-expert token lists + shared gate.
__global__ __launch_bounds__(64) void router_kernel(
    const float* __restrict__ x, const float* __restrict__ rw,
    const float* __restrict__ egw, int* __restrict__ counts,
    int* __restrict__ tlist, float* __restrict__ wlist,
    float* __restrict__ shgate) {
  __shared__ float xs[kH];
  __shared__ float logits[kE];
  const int t = blockIdx.x;
  const int tid = threadIdx.x;
  const float* xrow = x + (size_t)t * kH;
  for (int i = tid; i < kH; i += 64) xs[i] = xrow[i];
  __syncthreads();
  if (tid < kE) {
    const float* wr = rw + (size_t)tid * kH;
    float acc = 0.0f;
    for (int k = 0; k < kH; ++k) acc += xs[k] * wr[k];
    logits[tid] = acc;
  } else if (tid == kE) {
    float acc = 0.0f;
    for (int k = 0; k < kH; ++k) acc += xs[k] * egw[k];
    shgate[t] = fast_sigmoid(acc);
  }
  __syncthreads();
  if (tid == 0) {
    float mx = -3.4e38f;
    for (int e = 0; e < kE; ++e) mx = fmaxf(mx, logits[e]);
    float s = 0.0f;
    for (int e = 0; e < kE; ++e) {
      float p = __expf(logits[e] - mx);
      logits[e] = p;
      s += p;
    }
    const float inv = __builtin_amdgcn_rcpf(s);
    for (int k = 0; k < kTopK; ++k) {   // NORM_TOPK_PROB=False: raw probs
      int be = 0;
      float bv = -1.0f;
      for (int e = 0; e < kE; ++e)
        if (logits[e] > bv) { bv = logits[e]; be = e; }
      logits[be] = -2.0f;
      int pos = atomicAdd(&counts[be], 1);
      tlist[be * kT + pos] = t;
      wlist[be * kT + pos] = bv * inv;
    }
  }
}

// ---------------------------------------------------------------------------
// Blocks 0..31: shared expert (I=5632, chunked by 1408), token tile = bx.
// Blocks 32.. : routed experts, one workgroup per (expert, 16-token tile).
__global__ __launch_bounds__(kThreads) void moe_expert_kernel(
    const float* __restrict__ x,
    const float* __restrict__ gate_w, const float* __restrict__ up_w,
    const float* __restrict__ down_w,
    const float* __restrict__ sh_gate_w, const float* __restrict__ sh_up_w,
    const float* __restrict__ sh_down_w,
    const int* __restrict__ counts, const int* __restrict__ tlist,
    const float* __restrict__ wlist, const float* __restrict__ shgate,
    float* __restrict__ out) {
  extern __shared__ unsigned short smem_us[];
  unsigned short* Xh = smem_us;                 // [16][kXStrH]
  unsigned short* Xl = Xh + 16 * kXStrH;        // [16][kXStrH]
  unsigned short* Ah = Xl + 16 * kXStrH;        // [16][kAStrH]
  unsigned short* Al = Ah + 16 * kAStrH;        // [16][kAStrH]
  __shared__ int   toks[16];
  __shared__ float twts[16];

  const int bx = blockIdx.x;
  const int tid = threadIdx.x;
  const bool is_shared = bx < 32;               // fat blocks scheduled first

  int e = 0, tile, Itot, rowlen;
  const float *gw, *uw, *dw;
  if (is_shared) {
    tile = bx;
    gw = sh_gate_w; uw = sh_up_w; dw = sh_down_w;
    Itot = kSI; rowlen = kSI;
  } else {
    const int b2 = bx - 32;
    e = b2 >> 5;
    tile = b2 & 31;
    if (tile * 16 >= counts[e]) return;         // uniform early-exit
    gw = gate_w + (size_t)e * kI * kH;
    uw = up_w   + (size_t)e * kI * kH;
    dw = down_w + (size_t)e * kH * kI;
    Itot = kI; rowlen = kI;
  }

  if (tid < 16) {
    int r = tile * 16 + tid;
    if (is_shared) {
      toks[tid] = r;
      twts[tid] = shgate[r];
    } else {
      int cnt = counts[e];
      if (r < cnt) { toks[tid] = tlist[e * kT + r]; twts[tid] = wlist[e * kT + r]; }
      else         { toks[tid] = 0;                 twts[tid] = 0.0f; }
    }
  }
  __syncthreads();

  // Stage X tile into LDS, pre-split to bf16 hi/lo (split cost paid once).
  for (int i = tid; i < 16 * (kH / 4); i += kThreads) {
    int r = i >> 9;               // kH/4 = 512
    int c = (i & 511) * 4;
    v4f xv = *(const v4f*)(x + (size_t)toks[r] * kH + c);
    unsigned h0, l0, h1, l1;
    split2(xv[0], xv[1], h0, l0);
    split2(xv[2], xv[3], h1, l1);
    v2u hh = {h0, h1}, ll = {l0, l1};
    *(v2u*)(Xh + r * kXStrH + c) = hh;
    *(v2u*)(Xl + r * kXStrH + c) = ll;
  }
  __syncthreads();

  const int wave = tid >> 5;
  const int lane = tid & 31;
  const int lm = lane & 15;       // M (A/C row) or N (B/C col)
  const int lh = lane >> 4;       // lane half -> K sub-range

  v8f zero8 = {};
  v8f acc[kNT];
#pragma unroll
  for (int n = 0; n < kNT; ++n) acc[n] = zero8;

  for (int c0 = 0; c0 < Itot; c0 += kChunk) {
    // ---- Stage A: act = silu(X@gw^T) * (X@uw^T), 16 x kChunk, bf16x2 ------
    for (int it = wave; it < kChunk / 16; it += kWaves) {
      const int ibase = c0 + it * 16;
      const float* grow = gw + (size_t)(ibase + lm) * kH + 16 * lh;  // B rows
      const float* urow = uw + (size_t)(ibase + lm) * kH + 16 * lh;
      const unsigned short* xh = Xh + lm * kXStrH + 16 * lh;         // A rows
      const unsigned short* xl = Xl + lm * kXStrH + 16 * lh;
      v8f g = zero8, u = zero8;
      for (int k0 = 0; k0 < kH; k0 += 32) {
        BV ah, al;
        *(v4u*)&ah.u[0] = *(const v4u*)(xh + k0);        // A: K=k0+16h..+7
        *(v4u*)&ah.u[4] = *(const v4u*)(xh + k0 + 8);    //    K=k0+16h+8..+15
        *(v4u*)&al.u[0] = *(const v4u*)(xl + k0);
        *(v4u*)&al.u[4] = *(const v4u*)(xl + k0 + 8);
        BV gh, gl, uh, ul;
        split16(grow + k0, gh, gl);                      // B: K=k0+16h..+15
        split16(urow + k0, uh, ul);
        g = wmma_bf(ah.v, gh.v, g);
        g = wmma_bf(ah.v, gl.v, g);
        g = wmma_bf(al.v, gh.v, g);
        u = wmma_bf(ah.v, uh.v, u);
        u = wmma_bf(ah.v, ul.v, u);
        u = wmma_bf(al.v, uh.v, u);
      }
      // C layout: VGPR r, lanes0-15 -> M=r, lanes16-31 -> M=r+8; N=lm.
#pragma unroll
      for (int r = 0; r < 8; ++r) {
        float gv = g[r];
        float av = gv * fast_sigmoid(gv) * u[r];   // silu(g)*u, no IEEE div
        unsigned uv = __float_as_uint(av);
        float rem = av - __uint_as_float(uv & 0xffff0000u);
        const int off = (r + 8 * lh) * kAStrH + it * 16 + lm;
        Ah[off] = (unsigned short)(uv >> 16);
        Al[off] = (unsigned short)(__float_as_uint(rem) >> 16);
      }
    }
    __syncthreads();
    // ---- Stage B: acc += act @ down^T (this wave's 128 output cols) -------
    {
      const unsigned short* ph = Ah + lm * kAStrH + 16 * lh;
      const unsigned short* pl = Al + lm * kAStrH + 16 * lh;
      for (int k0 = 0; k0 < kChunk; k0 += 32) {
        BV ah, al;
        *(v4u*)&ah.u[0] = *(const v4u*)(ph + k0);
        *(v4u*)&ah.u[4] = *(const v4u*)(ph + k0 + 8);
        *(v4u*)&al.u[0] = *(const v4u*)(pl + k0);
        *(v4u*)&al.u[4] = *(const v4u*)(pl + k0 + 8);
#pragma unroll
        for (int nt = 0; nt < kNT; ++nt) {
          const float* drow =
              dw + (size_t)(wave * (16 * kNT) + nt * 16 + lm) * rowlen +
              c0 + k0 + 16 * lh;
          BV bh, bl;
          split16(drow, bh, bl);
          acc[nt] = wmma_bf(ah.v, bh.v, acc[nt]);
          acc[nt] = wmma_bf(ah.v, bl.v, acc[nt]);
          acc[nt] = wmma_bf(al.v, bh.v, acc[nt]);
        }
      }
    }
    __syncthreads();   // protect act buffer before next chunk (shared expert)
  }

  // Epilogue: scale by routing / sigmoid-shared weight, combine via atomics.
#pragma unroll
  for (int nt = 0; nt < kNT; ++nt) {
    const int col = wave * (16 * kNT) + nt * 16 + lm;
#pragma unroll
    for (int r = 0; r < 8; ++r) {
      const int m = r + 8 * lh;
      float v = acc[nt][r] * twts[m];
      atomicAdd(out + (size_t)toks[m] * kH + col, v);
    }
  }
}

// ---------------------------------------------------------------------------
extern "C" void kernel_launch(void* const* d_in, const int* in_sizes, int n_in,
                              void* d_out, int out_size, void* d_ws,
                              size_t ws_size, hipStream_t stream) {
  const float* x         = (const float*)d_in[0];
  const float* router_w  = (const float*)d_in[1];
  const float* gate_w    = (const float*)d_in[2];
  const float* up_w      = (const float*)d_in[3];
  const float* down_w    = (const float*)d_in[4];
  const float* sh_gate_w = (const float*)d_in[5];
  const float* sh_up_w   = (const float*)d_in[6];
  const float* sh_down_w = (const float*)d_in[7];
  const float* egw       = (const float*)d_in[8];
  // d_in[9] = top_k (known constant 4)
  float* out = (float*)d_out;

  // Workspace layout (4-byte units)
  int*   counts = (int*)d_ws;                 // [64] (kE used)
  int*   tlist  = counts + 64;                // [kE*kT]
  float* wlist  = (float*)(tlist + kE * kT);  // [kE*kT]
  float* shgate = wlist + kE * kT;            // [kT]

  (void)hipFuncSetAttribute((const void*)moe_expert_kernel,
                            hipFuncAttributeMaxDynamicSharedMemorySize,
                            kSmemBytes);

  zero_kernel<<<(kT * kH + 255) / 256, 256, 0, stream>>>(out, counts);
  router_kernel<<<kT, 64, 0, stream>>>(x, router_w, egw, counts, tlist, wlist,
                                       shgate);
  moe_expert_kernel<<<32 + kE * 32, kThreads, kSmemBytes, stream>>>(
      x, gate_w, up_w, down_w, sh_gate_w, sh_up_w, sh_down_w, counts, tlist,
      wlist, shgate, out);
}
